// LSTM_Autoencoder_32753420599734
// MI455X (gfx1250) — matrix operand order repaired and checked by reading, not compiled, not run on hardware
//
#include <hip/hip_runtime.h>
#include <hip/hip_bf16.h>
#include <cstddef>

// ---------------------------------------------------------------------------
// Fused LSTM autoencoder for MI455X (gfx1250, wave32, WMMA).
// One wave = one 16-row batch tile, resident for all 256 timesteps.
// Gate GEMMs: v_wmma_f32_16x16x32_bf16 (bf16 A/B, f32 accumulate).
// All h/c state + weights live in LDS/registers; HBM traffic = x + out only.
// Round 3: sigmoid via native v_tanh_f32 (0.5*tanh(0.5x)+0.5) -> one trans op
// per gate instead of exp+rcp, shortening the recurrence critical path.
// ---------------------------------------------------------------------------

typedef __attribute__((ext_vector_type(16))) __bf16 v16bf;
typedef __attribute__((ext_vector_type(8)))  __bf16 v8bf;
typedef __attribute__((ext_vector_type(8)))  float  v8f;
typedef __attribute__((ext_vector_type(4)))  float  v4f;

#define WAVES 2
#define ACT_PER_WAVE 4096   // bf16 elements per wave

// per-wave activation buffer offsets (bf16 elems), each tile is 16 rows
#define A_E1 0       // 16 x 64  (x:0..31, h_e1:32..47, pad)
#define A_E2 1024    // 16 x 32  (h_e1:0..15, h_e2:16..19, pad)
#define A_E3 1536    // 16 x 32  (h_e2:0..3,  h_e3:4, pad)
#define A_D1 2048    // 16 x 32  (enc:0,      h_d1:1..4, pad)
#define A_D2 2560    // 16 x 32  (h_d1:0..3,  h_d2:4..11, pad)
#define A_D3 3072    // 16 x 32  (h_d2:0..7,  h_d3:8..23, pad)
#define A_OUT 3584   // 16 x 32  (h_d3:0..15, pad)

// shared weight offsets (bf16 elems), row-major [Npad][Kp]
#define W_E1 0       // 64 x 64
#define W_E2 4096    // 16 x 32
#define W_E3 4608    // 16 x 32
#define W_D1 5120    // 16 x 32
#define W_D2 5632    // 32 x 32
#define W_D3 6656    // 64 x 32
#define W_OUT 8704   // 32 x 32
#define W_TOT 9728

struct Params {
    const float* x;
    const float* Wih[6];
    const float* Whh[6];
    const float* bih[6];
    const float* bhh[6];
    const float* Wout;
    const float* bout;
    float*       out;
};

// --- fast activations on the gfx1250 transcendental pipe ---
__device__ __forceinline__ float tanh_f(float x) {
#if __has_builtin(__builtin_amdgcn_tanhf)
    return __builtin_amdgcn_tanhf(x);          // native v_tanh_f32
#elif __has_builtin(__builtin_amdgcn_tanh_f32)
    return __builtin_amdgcn_tanh_f32(x);
#else
    return 2.0f * __builtin_amdgcn_rcpf(1.0f + __expf(-2.0f * x)) - 1.0f;
#endif
}
__device__ __forceinline__ float sigm(float x) {
#if __has_builtin(__builtin_amdgcn_tanhf) || __has_builtin(__builtin_amdgcn_tanh_f32)
    // sigmoid(x) = 0.5*tanh(x/2) + 0.5 : one v_tanh_f32 + mul + fma
    return __builtin_fmaf(0.5f, tanh_f(0.5f * x), 0.5f);
#else
    return __builtin_amdgcn_rcpf(1.0f + __expf(-x));
#endif
}

// A-operand (16x32 bf16): lane<16 holds K{kh..kh+7, kh+16..kh+23}, kh=0;
// lane>=16: kh=8.  Two ds_read_b128 per tile.
__device__ __forceinline__ v16bf lds_tile_a(const __bf16* base, int stride,
                                            int kc, int lane) {
    int row = lane & 15;
    int kh  = (lane >> 4) << 3;
    const __bf16* p = base + row * stride + kc * 32 + kh;
    v8bf lo = *(const v8bf*)p;
    v8bf hi = *(const v8bf*)(p + 16);
    v16bf r;
#pragma unroll
    for (int i = 0; i < 8; ++i) { r[i] = lo[i]; r[i + 8] = hi[i]; }
    return r;
}

// B-operand (32x16 bf16) from W stored row-major [N][K]: lane n holds
// K{kh..kh+7, kh+16..kh+23}, kh = (lane>=16 ? 8 : 0).
__device__ __forceinline__ v16bf lds_tile_b(const __bf16* base, int stride,
                                            int kc, int lane) {
    int row = lane & 15;
    int ko  = (lane >> 4) << 3;
    const __bf16* p = base + row * stride + kc * 32 + ko;
    v8bf lo = *(const v8bf*)p;
    v8bf hi = *(const v8bf*)(p + 16);
    v16bf r;
#pragma unroll
    for (int i = 0; i < 8; ++i) { r[i] = lo[i]; r[i + 8] = hi[i]; }
    return r;
}

// gates[16 x NT*16] = act[16 x KP] @ Wcat^T  (+bias splat per N column)
template <int KP, int NT>
__device__ __forceinline__ void gemm(const __bf16* act, int astride,
                                     const __bf16* w, const float* bias,
                                     v8f* C, int lane) {
    v16bf a[KP / 32];
#pragma unroll
    for (int kc = 0; kc < KP / 32; ++kc) a[kc] = lds_tile_a(act, astride, kc, lane);
#pragma unroll
    for (int nt = 0; nt < NT; ++nt) {
        v8f acc;
        float b = bias[nt];
#pragma unroll
        for (int i = 0; i < 8; ++i) acc[i] = b;
#pragma unroll
        for (int kc = 0; kc < KP / 32; ++kc) {
            v16bf bb = lds_tile_b(w + nt * 16 * KP, KP, kc, lane);
            acc = __builtin_amdgcn_wmma_f32_16x16x32_bf16(
                false, a[kc], false, bb, (short)0, acc, false, false);
        }
        C[nt] = acc;
    }
}

// H==16: gate type == N-tile, same lane/row coords -> no cross-lane traffic.
__device__ __forceinline__ void lstm_ew16(const v8f* C, float* cst,
                                          __bf16* rec, int rstride, int rcol,
                                          __bf16* nxt, int ncol, int lane) {
    int h  = lane & 15;
    int mr = (lane >> 4) * 8;
#pragma unroll
    for (int r = 0; r < 8; ++r) {
        int m = mr + r;
        float i_ = sigm(C[0][r]);
        float f_ = sigm(C[1][r]);
        float g_ = tanh_f(C[2][r]);
        float o_ = sigm(C[3][r]);
        float c = f_ * cst[r] + i_ * g_;
        cst[r] = c;
        float hv = o_ * tanh_f(c);
        __bf16 hb = (__bf16)hv;
        rec[m * rstride + rcol + h] = hb;
        nxt[m * 32 + ncol + h]      = hb;
    }
}

// H<16: round-trip gates through per-wave LDS f32 buffer (stride 32).
template <int H, int NT>
__device__ __forceinline__ void lstm_small(const v8f* C, float* sg, float* cst,
                                           __bf16* rec, int rcol,
                                           __bf16* nxt, int ncol, int lane) {
    int n0 = lane & 15;
    int mr = (lane >> 4) * 8;
#pragma unroll
    for (int nt = 0; nt < NT; ++nt) {
#pragma unroll
        for (int r = 0; r < 8; ++r)
            sg[(mr + r) * 32 + nt * 16 + n0] = C[nt][r];
    }
    int m   = lane & 15;
    int hb2 = (lane >> 4);
#pragma unroll
    for (int j = 0; j < (H + 1) / 2; ++j) {
        int h = hb2 + 2 * j;
        if (h < H) {
            const float* g = sg + m * 32;
            float i_ = sigm(g[h]);
            float f_ = sigm(g[H + h]);
            float gg = tanh_f(g[2 * H + h]);
            float o_ = sigm(g[3 * H + h]);
            float c = f_ * cst[j] + i_ * gg;
            cst[j] = c;
            float hv = o_ * tanh_f(c);
            __bf16 hbv = (__bf16)hv;
            rec[m * 32 + rcol + h] = hbv;
            nxt[m * 32 + ncol + h] = hbv;
        }
    }
}

__global__ __launch_bounds__(32 * WAVES)
void lstm_ae_fused(Params p) {
    __shared__ __bf16 sW[W_TOT];
    __shared__ __bf16 sActAll[WAVES * ACT_PER_WAVE];
    __shared__ float  sGAll[WAVES * 512];

    const int tid  = threadIdx.x;
    const int lane = tid & 31;
    const int wave = tid >> 5;

    // ---- cooperative weight fill: Wcat[n][k] = [Wih | Whh], bf16, zero pad
    {
        const int Din[6]  = {32, 16, 4, 1, 4, 8};
        const int Hh[6]   = {16, 4, 1, 4, 8, 16};
        const int Kp[6]   = {64, 32, 32, 32, 32, 32};
        const int Npad[6] = {64, 16, 16, 16, 32, 64};
        const int Woff[6] = {W_E1, W_E2, W_E3, W_D1, W_D2, W_D3};
#pragma unroll
        for (int l = 0; l < 6; ++l) {
            int tot = Npad[l] * Kp[l];
            for (int idx = tid; idx < tot; idx += blockDim.x) {
                int n = idx / Kp[l];
                int k = idx % Kp[l];
                float v = 0.0f;
                if (n < 4 * Hh[l]) {
                    if (k < Din[l])              v = p.Wih[l][n * Din[l] + k];
                    else if (k < Din[l] + Hh[l]) v = p.Whh[l][n * Hh[l] + (k - Din[l])];
                }
                sW[Woff[l] + idx] = (__bf16)v;
            }
        }
        for (int idx = tid; idx < 1024; idx += blockDim.x) {
            int n = idx / 32, k = idx % 32;
            float v = (k < 16) ? p.Wout[n * 16 + k] : 0.0f;
            sW[W_OUT + idx] = (__bf16)v;
        }
        for (int idx = tid; idx < WAVES * ACT_PER_WAVE; idx += blockDim.x)
            sActAll[idx] = (__bf16)0.0f;
    }
    __syncthreads();

    __bf16* act = sActAll + wave * ACT_PER_WAVE;
    float*  sg  = sGAll + wave * 512;
    const int b0 = blockIdx.x * (WAVES * 16) + wave * 16;

    // ---- per-lane bias registers: bias[n] = bih[n]+bhh[n], n = nt*16 + lane&15
    const int n0 = lane & 15;
    float b_e1[4], b_e2[1], b_e3[1], b_d1[1], b_d2[2], b_d3[4], b_o[2];
#pragma unroll
    for (int nt = 0; nt < 4; ++nt) {
        int n = nt * 16 + n0;
        b_e1[nt] = p.bih[0][n] + p.bhh[0][n];
        b_d3[nt] = p.bih[5][n] + p.bhh[5][n];
    }
    b_e2[0] = p.bih[1][n0] + p.bhh[1][n0];
    b_e3[0] = (n0 < 4) ? (p.bih[2][n0] + p.bhh[2][n0]) : 0.0f;
    b_d1[0] = p.bih[3][n0] + p.bhh[3][n0];
#pragma unroll
    for (int nt = 0; nt < 2; ++nt) {
        int n = nt * 16 + n0;
        b_d2[nt] = p.bih[4][n] + p.bhh[4][n];
        b_o[nt]  = p.bout[n];
    }

    // cell states (f32, resident in VGPRs for all 128 steps)
    float c_e1[8] = {}, c_e2[2] = {}, c_e3[1] = {}, c_d1[2] = {}, c_d2[4] = {},
          c_d3[8] = {};
    v8f C[4];

    // ------------------------------ encoder ------------------------------
    for (int t = 0; t < 128; ++t) {
        // stage x[b0..b0+15][t][0..31] into act_e1 cols 0..31 as bf16
        {
            int m  = lane & 15;
            int f0 = (lane >> 4) * 16;
            const float* xp =
                p.x + ((size_t)(b0 + m) * 128 + (size_t)t) * 32 + f0;
            v4f a0 = *(const v4f*)xp;
            v4f a1 = *(const v4f*)(xp + 4);
            v4f a2 = *(const v4f*)(xp + 8);
            v4f a3 = *(const v4f*)(xp + 12);
            if (t < 127) __builtin_prefetch(xp + 32, 0, 1);
            v8bf d0, d1;
#pragma unroll
            for (int i = 0; i < 4; ++i) {
                d0[i]     = (__bf16)a0[i];
                d0[4 + i] = (__bf16)a1[i];
                d1[i]     = (__bf16)a2[i];
                d1[4 + i] = (__bf16)a3[i];
            }
            __bf16* dst = act + A_E1 + m * 64 + f0;
            *(v8bf*)dst       = d0;
            *(v8bf*)(dst + 8) = d1;
        }
        // e1: Din 32, H 16 -> Kp 64, NT 4
        gemm<64, 4>(act + A_E1, 64, sW + W_E1, b_e1, C, lane);
        lstm_ew16(C, c_e1, act + A_E1, 64, 32, act + A_E2, 0, lane);
        // e2: Din 16, H 4
        gemm<32, 1>(act + A_E2, 32, sW + W_E2, b_e2, C, lane);
        lstm_small<4, 1>(C, sg, c_e2, act + A_E2, 16, act + A_E3, 0, lane);
        // e3: Din 4, H 1; h feeds its own recurrence + enc slot (d1 col 0)
        gemm<32, 1>(act + A_E3, 32, sW + W_E3, b_e3, C, lane);
        lstm_small<1, 1>(C, sg, c_e3, act + A_E3, 4, act + A_D1, 0, lane);
    }

    // ------------------------------ decoder ------------------------------
    for (int t = 0; t < 128; ++t) {
        // d1: Din 1 (enc, constant), H 4
        gemm<32, 1>(act + A_D1, 32, sW + W_D1, b_d1, C, lane);
        lstm_small<4, 1>(C, sg, c_d1, act + A_D1, 1, act + A_D2, 0, lane);
        // d2: Din 4, H 8
        gemm<32, 2>(act + A_D2, 32, sW + W_D2, b_d2, C, lane);
        lstm_small<8, 2>(C, sg, c_d2, act + A_D2, 4, act + A_D3, 0, lane);
        // d3: Din 8, H 16
        gemm<32, 4>(act + A_D3, 32, sW + W_D3, b_d3, C, lane);
        lstm_ew16(C, c_d3, act + A_D3, 32, 8, act + A_OUT, 0, lane);
        // output projection: h6[16x16] @ Wout^T[16x32] + bout
        gemm<32, 2>(act + A_OUT, 32, sW + W_OUT, b_o, C, lane);
        {
            int mr = (lane >> 4) * 8;
#pragma unroll
            for (int nt = 0; nt < 2; ++nt) {
#pragma unroll
                for (int r = 0; r < 8; ++r) {
                    int m = mr + r;
                    int f = nt * 16 + n0;
                    p.out[((size_t)(b0 + m) * 128 + (size_t)t) * 32 + f] =
                        C[nt][r];
                }
            }
        }
    }
}

extern "C" void kernel_launch(void* const* d_in, const int* in_sizes, int n_in,
                              void* d_out, int out_size, void* d_ws,
                              size_t ws_size, hipStream_t stream) {
    (void)in_sizes; (void)n_in; (void)out_size; (void)d_ws; (void)ws_size;
    Params p;
    p.x = (const float*)d_in[0];
    for (int l = 0; l < 6; ++l) {
        p.Wih[l] = (const float*)d_in[1 + 4 * l];
        p.Whh[l] = (const float*)d_in[2 + 4 * l];
        p.bih[l] = (const float*)d_in[3 + 4 * l];
        p.bhh[l] = (const float*)d_in[4 + 4 * l];
    }
    p.Wout = (const float*)d_in[25];
    p.bout = (const float*)d_in[26];
    p.out  = (float*)d_out;

    dim3 grid(4096 / (16 * WAVES));  // 128 blocks
    dim3 block(32 * WAVES);          // 2 waves, each owns a 16-row tile
    lstm_ae_fused<<<grid, block, 0, stream>>>(p);
}
